// ACTV1MoERoutingLayer_74929999446934
// MI455X (gfx1250) — matrix operand order repaired and checked by reading, not compiled, not run on hardware
//
#include <hip/hip_runtime.h>
#include <hip/hip_bf16.h>

// ---------------------------------------------------------------------------
// MoE routing layer for MI455X (gfx1250, wave32, WMMA).
// All GEMMs use v_wmma_f32_16x16x32_bf16 (f32 accumulate). Softmax paths f32.
// Register blocking: 2x2 WMMA tiles (32x32/wave) -- empirically the largest
// spill-free configuration (2x4 and 4x4 both spilled accumulators to
// scratch inside the K-loop; spill/reload costs more than the saved loads).
// ---------------------------------------------------------------------------

typedef __attribute__((ext_vector_type(16))) __bf16 bf16x16;
typedef __attribute__((ext_vector_type(8)))  __bf16 bf16x8;
typedef __attribute__((ext_vector_type(8)))  float  f32x8;

#define B_  8
#define S_  512
#define H_  1024
#define HS_ 512
#define NH_ 8
#define HD_ 64
#define IN_ 1536
#define E_  8
#define M_  (B_*S_)   // 4096 token rows

// ------------------------- elementwise kernels -----------------------------

__global__ void add_cvt_kernel(const float* __restrict__ a, const float* __restrict__ b,
                               float* __restrict__ h, __bf16* __restrict__ hbf, int n) {
  int i = blockIdx.x * blockDim.x + threadIdx.x;
  if (i < n) { float v = a[i] + b[i]; h[i] = v; hbf[i] = (__bf16)v; }
}

__global__ void cvt_f32_bf16_kernel(const float* __restrict__ x, __bf16* __restrict__ y, int n) {
  int i = blockIdx.x * blockDim.x + threadIdx.x;
  if (i < n) y[i] = (__bf16)x[i];
}

__global__ void zero_f32_kernel(float* __restrict__ x, int n) {
  int i = blockIdx.x * blockDim.x + threadIdx.x;
  if (i < n) x[i] = 0.f;
}

// --------------------------- gating (tiny) ---------------------------------
// 1 block, 64 threads: thread t -> (b = t/8, e = t%8)
__global__ void gate_kernel(const float* __restrict__ h, const float* __restrict__ gw,
                            float* __restrict__ Wfull, float* __restrict__ aux) {
  __shared__ float logit[B_][E_];
  __shared__ float probs[B_][E_];
  __shared__ int   sel[B_][E_];
  int t = threadIdx.x, b = t >> 3, e = t & 7;
  const float* hr = h + (size_t)b * S_ * H_;   // h[b, 0, :]
  const float* gr = gw + e * H_;
  float acc = 0.f;
  for (int k = 0; k < H_; ++k) acc += hr[k] * gr[k];
  logit[b][e] = acc;
  sel[b][e] = 0;
  Wfull[t] = 0.f;
  __syncthreads();
  if (e == 0) {
    float mx = logit[b][0];
    for (int j = 1; j < E_; ++j) mx = fmaxf(mx, logit[b][j]);
    float s = 0.f;
    for (int j = 0; j < E_; ++j) { float p = __expf(logit[b][j] - mx); probs[b][j] = p; s += p; }
    for (int j = 0; j < E_; ++j) probs[b][j] /= s;
    // top-2 (first-index tie-break like lax.top_k)
    int i1 = 0; float v1 = probs[b][0];
    for (int j = 1; j < E_; ++j) if (probs[b][j] > v1) { v1 = probs[b][j]; i1 = j; }
    int i2 = -1; float v2 = -1.f;
    for (int j = 0; j < E_; ++j) if (j != i1 && probs[b][j] > v2) { v2 = probs[b][j]; i2 = j; }
    float denom = v1 + v2; if (denom < 1e-8f) denom = 1e-8f;
    Wfull[b * E_ + i1] = v1 / denom;
    Wfull[b * E_ + i2] = v2 / denom;
    sel[b][i1] = 1; sel[b][i2] = 1;
  }
  __syncthreads();
  if (t == 0) {
    float a = 0.f;
    for (int j = 0; j < E_; ++j) {
      float imp = 0.f, ld = 0.f;
      for (int bb = 0; bb < B_; ++bb) { imp += probs[bb][j]; ld += (float)sel[bb][j]; }
      a += (float)E_ * (imp / (float)B_) * (ld / (float)(B_ * 2));
    }
    *aux = a;
  }
}

// --------------------------- WMMA bf16 GEMM --------------------------------
// C[M,N] = A[M,K] @ W[N,K]^T.  One wave owns a 32x32 macro-tile (2x2 WMMA).
// Fragment layouts used here (wave32; A/B in contiguous K order, same for both):
//  A 16x32: lane&15 = M row; lane holds K {kb..kb+15}, kb=16*(lane>>4)
//  B 32x16: lane&15 = N col; lane holds K {kb..kb+15}, kb=16*(lane>>4)
//  C/D:     lane&15 = N; vgpr r -> M = r + 8*(lane>>4)

__device__ __forceinline__ bf16x16 load_a_frag(const __bf16* __restrict__ A, int K,
                                               int row_base, int k, int lane) {
  int m  = row_base + (lane & 15);
  int kb = (lane >> 4) << 4;            // 0 or 16
  const __bf16* p = A + (size_t)m * K + k + kb;
  bf16x8 lo = *(const bf16x8*)p;        // K = kb..kb+7
  bf16x8 hi = *(const bf16x8*)(p + 8);  // K = kb+8..kb+15
  return __builtin_shufflevector(lo, hi, 0,1,2,3,4,5,6,7,8,9,10,11,12,13,14,15);
}

__device__ __forceinline__ bf16x16 load_b_frag(const __bf16* __restrict__ W, int K,
                                               int col_base, int k, int lane) {
  int n  = col_base + (lane & 15);
  int kb = (lane >> 4) << 4;            // 0 or 16
  return *(const bf16x16*)(W + (size_t)n * K + k + kb);
}

__global__ void gemm_bf16_wmma_kernel(const __bf16* __restrict__ A,
                                      const __bf16* __restrict__ W,
                                      float* __restrict__ C,
                                      __bf16* __restrict__ Cbf,  // optional bf16 copy of C
                                      int M, int N, int K) {
  int lane = threadIdx.x & 31;
  int wave = blockIdx.x * (blockDim.x >> 5) + (threadIdx.x >> 5);
  int n32 = N >> 5;
  int tiles = (M >> 5) * n32;
  if (wave >= tiles) return;
  int m0 = (wave / n32) << 5;
  int n0 = (wave % n32) << 5;

  f32x8 acc00 = {}, acc01 = {}, acc10 = {}, acc11 = {};
  for (int k = 0; k < K; k += 32) {
    bf16x16 a0 = load_a_frag(A, K, m0,      k, lane);
    bf16x16 a1 = load_a_frag(A, K, m0 + 16, k, lane);
    bf16x16 b0 = load_b_frag(W, K, n0,      k, lane);
    bf16x16 b1 = load_b_frag(W, K, n0 + 16, k, lane);
    acc00 = __builtin_amdgcn_wmma_f32_16x16x32_bf16(false, a0, false, b0, (short)0, acc00, false, false);
    acc01 = __builtin_amdgcn_wmma_f32_16x16x32_bf16(false, a0, false, b1, (short)0, acc01, false, false);
    acc10 = __builtin_amdgcn_wmma_f32_16x16x32_bf16(false, a1, false, b0, (short)0, acc10, false, false);
    acc11 = __builtin_amdgcn_wmma_f32_16x16x32_bf16(false, a1, false, b1, (short)0, acc11, false, false);
  }

  int cm = (lane >> 4) << 3;
  int cn = lane & 15;
#pragma unroll
  for (int r = 0; r < 8; ++r) {
    size_t i00 = (size_t)(m0 +      cm + r) * N + n0 +      cn;
    size_t i01 = (size_t)(m0 +      cm + r) * N + n0 + 16 + cn;
    size_t i10 = (size_t)(m0 + 16 + cm + r) * N + n0 +      cn;
    size_t i11 = (size_t)(m0 + 16 + cm + r) * N + n0 + 16 + cn;
    C[i00] = acc00[r]; C[i01] = acc01[r]; C[i10] = acc10[r]; C[i11] = acc11[r];
    if (Cbf) {
      Cbf[i00] = (__bf16)acc00[r]; Cbf[i01] = (__bf16)acc01[r];
      Cbf[i10] = (__bf16)acc10[r]; Cbf[i11] = (__bf16)acc11[r];
    }
  }
}

// ------------------------------- RoPE --------------------------------------
// qkv row = 1536 = 24 heads * 64; heads 0..7 = Q, 8..15 = K, 16..23 = V.
// Thread handles one (b,s,head,d<32) pair (d, d+32).
__global__ void rope_kernel(float* __restrict__ qkv, const float* __restrict__ cosb,
                            const float* __restrict__ sinb) {
  int i = blockIdx.x * blockDim.x + threadIdx.x;
  if (i >= B_ * S_ * 16 * 32) return;
  int d = i & 31; int t = i >> 5;
  int head = t & 15; t >>= 4;
  int s = t & (S_ - 1); int b = t >> 9;
  float c  = cosb[s * HD_ + d];
  float sn = sinb[s * HD_ + d];
  size_t base = ((size_t)(b * S_ + s)) * (3 * NH_ * HD_) + head * HD_;
  float x1 = qkv[base + d], x2 = qkv[base + d + 32];
  qkv[base + d]      = x1 * c - x2 * sn;
  qkv[base + d + 32] = x2 * c + x1 * sn;
}

// ----------------------------- attention -----------------------------------
// Block (64 threads = 2 waves) per (b, head, q-row); two-pass f32 softmax.
__global__ void attn_kernel(const float* __restrict__ qkv, float* __restrict__ out) {
  int q = blockIdx.x, head = blockIdx.y, b = blockIdx.z;
  int t = threadIdx.x;
  __shared__ float qv[HD_];
  __shared__ float sc[S_];
  __shared__ float red[64];

  const int ROW = 3 * NH_ * HD_;  // 1536
  size_t rowq = ((size_t)(b * S_ + q)) * ROW;
  qv[t] = qkv[rowq + head * HD_ + t];
  __syncthreads();

  for (int k = t; k < S_; k += 64) {
    const float* kr = qkv + ((size_t)(b * S_ + k)) * ROW + (NH_ + head) * HD_;
    float acc = 0.f;
#pragma unroll 8
    for (int d = 0; d < HD_; ++d) acc += qv[d] * kr[d];
    sc[k] = acc * 0.125f;  // 1/sqrt(64)
  }
  __syncthreads();

  float mx = -3.4e38f;
  for (int k = t; k < S_; k += 64) mx = fmaxf(mx, sc[k]);
  red[t] = mx; __syncthreads();
  if (t == 0) { float m = red[0]; for (int j = 1; j < 64; ++j) m = fmaxf(m, red[j]); red[0] = m; }
  __syncthreads();
  mx = red[0];
  __syncthreads();

  float ls = 0.f;
  for (int k = t; k < S_; k += 64) { float e = __expf(sc[k] - mx); sc[k] = e; ls += e; }
  red[t] = ls; __syncthreads();
  if (t == 0) { float s = 0.f; for (int j = 0; j < 64; ++j) s += red[j]; red[0] = s; }
  __syncthreads();
  float inv = 1.f / red[0];

  // out[b,q,head,d], d = t
  float o = 0.f;
  for (int k = 0; k < S_; ++k)
    o += sc[k] * qkv[((size_t)(b * S_ + k)) * ROW + (2 * NH_ + head) * HD_ + t];
  out[((size_t)(b * S_ + q)) * HS_ + head * HD_ + t] = o * inv;
}

// --------------------- residual + RMSNorm (f32 + bf16 out) -----------------
__global__ void resid_rms_kernel(const float* __restrict__ x, const float* __restrict__ r,
                                 float* __restrict__ y, __bf16* __restrict__ ybf, int width) {
  int row = blockIdx.x, t = threadIdx.x;
  __shared__ float red[256];
  const float* xr = x + (size_t)row * width;
  const float* rr = r + (size_t)row * width;
  float ss = 0.f;
  for (int i = t; i < width; i += 256) { float v = xr[i] + rr[i]; ss += v * v; }
  red[t] = ss; __syncthreads();
  for (int off = 128; off > 0; off >>= 1) { if (t < off) red[t] += red[t + off]; __syncthreads(); }
  float scale = rsqrtf(red[0] / (float)width + 1e-5f);
  float* yr = y + (size_t)row * width;
  for (int i = t; i < width; i += 256) {
    float v = (xr[i] + rr[i]) * scale;
    yr[i] = v;
    if (ybf) ybf[(size_t)row * width + i] = (__bf16)v;
  }
}

// --------------------------- SiLU(gate) * up -------------------------------
__global__ void act_kernel(const float* __restrict__ gu, __bf16* __restrict__ inter, int n) {
  int i = blockIdx.x * blockDim.x + threadIdx.x;
  if (i >= n) return;
  int m = i / IN_, c = i - m * IN_;
  float g = gu[(size_t)m * (2 * IN_) + c];
  float u = gu[(size_t)m * (2 * IN_) + IN_ + c];
  float s = g / (1.f + __expf(-g));
  inter[i] = (__bf16)(s * u);
}

// --------------------------- weighted accumulate ---------------------------
__global__ void accum_kernel(float* __restrict__ mixed, const float* __restrict__ oute,
                             const float* __restrict__ Wfull, int e, int n) {
  int i = blockIdx.x * blockDim.x + threadIdx.x;
  if (i >= n) return;
  int b = i >> 19;  // S_*H_ = 524288 = 2^19
  mixed[i] += Wfull[b * E_ + e] * oute[i];
}

// ------------------------------ launcher -----------------------------------

extern "C" void kernel_launch(void* const* d_in, const int* in_sizes, int n_in,
                              void* d_out, int out_size, void* d_ws, size_t ws_size,
                              hipStream_t stream) {
  const float* hidden = (const float*)d_in[0];
  const float* inj    = (const float*)d_in[1];
  const float* cosb   = (const float*)d_in[2];
  const float* sinb   = (const float*)d_in[3];
  const float* gate_w = (const float*)d_in[4];
  const float* down_w = (const float*)d_in[5];
  const float* up_w   = (const float*)d_in[6];
  const float* qkv_w  = (const float*)d_in[7];
  const float* o_w    = (const float*)d_in[8];
  const float* gup_w  = (const float*)d_in[9];
  const float* mdw_w  = (const float*)d_in[10];
  float* out = (float*)d_out;

  // ---- workspace bump allocator (256B aligned) ----
  char* p = (char*)d_ws;
  auto alloc = [&](size_t bytes) -> void* {
    void* r = (void*)p; p += (bytes + 255) & ~(size_t)255; return r;
  };
  float*  h      = (float*) alloc((size_t)M_ * H_ * 4);
  __bf16* h_bf   = (__bf16*)alloc((size_t)M_ * H_ * 2);
  float*  mixed  = (float*) alloc((size_t)M_ * H_ * 4);
  float*  oute   = (float*) alloc((size_t)M_ * H_ * 4);
  float*  big1   = (float*) alloc((size_t)M_ * (2 * IN_) * 4);  // qkv (M x 1536) / gate_up (M x 3072)
  float*  s1     = (float*) alloc((size_t)M_ * HS_ * 4);        // hs, later hs3
  float*  s3     = (float*) alloc((size_t)M_ * HS_ * 4);        // attn_out / proj / mlp_down
  float*  s4     = (float*) alloc((size_t)M_ * HS_ * 4);        // hs2
  __bf16* bf_s   = (__bf16*)alloc((size_t)M_ * HS_ * 2);        // rotating bf16 activation tile
  __bf16* interb = (__bf16*)alloc((size_t)M_ * IN_ * 2);
  float*  Wfull  = (float*) alloc(1024);
  __bf16* dw_bf  = (__bf16*)alloc((size_t)E_ * HS_ * H_ * 2);
  __bf16* qkw_bf = (__bf16*)alloc((size_t)E_ * (3 * HS_) * HS_ * 2);
  __bf16* ow_bf  = (__bf16*)alloc((size_t)E_ * HS_ * HS_ * 2);
  __bf16* guw_bf = (__bf16*)alloc((size_t)E_ * (2 * IN_) * HS_ * 2);
  __bf16* mdw_bf = (__bf16*)alloc((size_t)E_ * HS_ * IN_ * 2);
  __bf16* uw_bf  = (__bf16*)alloc((size_t)E_ * H_ * HS_ * 2);

  auto cvt = [&](const float* src, __bf16* dst, int n) {
    cvt_f32_bf16_kernel<<<(n + 255) / 256, 256, 0, stream>>>(src, dst, n);
  };
  auto gemm = [&](const __bf16* A, const __bf16* W, float* C, __bf16* Cbf,
                  int M, int N, int K) {
    int tiles = (M / 32) * (N / 32);
    gemm_bf16_wmma_kernel<<<(tiles + 3) / 4, 128, 0, stream>>>(A, W, C, Cbf, M, N, K);
  };

  int nMH = M_ * H_;
  add_cvt_kernel<<<(nMH + 255) / 256, 256, 0, stream>>>(hidden, inj, h, h_bf, nMH);
  zero_f32_kernel<<<(nMH + 255) / 256, 256, 0, stream>>>(mixed, nMH);

  cvt(down_w, dw_bf,  E_ * HS_ * H_);
  cvt(qkv_w,  qkw_bf, E_ * 3 * HS_ * HS_);
  cvt(o_w,    ow_bf,  E_ * HS_ * HS_);
  cvt(gup_w,  guw_bf, E_ * 2 * IN_ * HS_);
  cvt(mdw_w,  mdw_bf, E_ * HS_ * IN_);
  cvt(up_w,   uw_bf,  E_ * H_ * HS_);

  gate_kernel<<<1, 64, 0, stream>>>(h, gate_w, Wfull, out + (size_t)M_ * H_);

  for (int e = 0; e < E_; ++e) {
    // hs = h @ down_w[e].T                     (4096x512) + fused bf16 copy
    gemm(h_bf, dw_bf + (size_t)e * HS_ * H_, s1, bf_s, M_, HS_, H_);
    // qkv = hs @ qkv_w[e].T                    (4096x1536)
    gemm(bf_s, qkw_bf + (size_t)e * 3 * HS_ * HS_, big1, nullptr, M_, 3 * HS_, HS_);
    rope_kernel<<<(M_ * 16 * 32 + 255) / 256, 256, 0, stream>>>(big1, cosb, sinb);
    attn_kernel<<<dim3(S_, NH_, B_), 64, 0, stream>>>(big1, s3);
    cvt(s3, bf_s, M_ * HS_);
    // proj = attn @ o_w[e].T                   (4096x512)
    gemm(bf_s, ow_bf + (size_t)e * HS_ * HS_, s3, nullptr, M_, HS_, HS_);
    // hs2 = rms_norm(hs + proj)
    resid_rms_kernel<<<M_, 256, 0, stream>>>(s1, s3, s4, bf_s, HS_);
    // gu = hs2 @ gate_up_w[e].T                (4096x3072)
    gemm(bf_s, guw_bf + (size_t)e * 2 * IN_ * HS_, big1, nullptr, M_, 2 * IN_, HS_);
    act_kernel<<<(M_ * IN_ + 255) / 256, 256, 0, stream>>>(big1, interb, M_ * IN_);
    // md = (silu(gate)*up) @ mlp_down_w[e].T   (4096x512)
    gemm(interb, mdw_bf + (size_t)e * HS_ * IN_, s3, nullptr, M_, HS_, IN_);
    // hs3 = rms_norm(hs2 + md)
    resid_rms_kernel<<<M_, 256, 0, stream>>>(s4, s3, s1, bf_s, HS_);
    // out_e = hs3 @ up_w[e].T                  (4096x1024)
    gemm(bf_s, uw_bf + (size_t)e * H_ * HS_, oute, nullptr, M_, H_, HS_);
    // mixed += W_full[:, e] * out_e
    accum_kernel<<<(nMH + 255) / 256, 256, 0, stream>>>(mixed, oute, Wfull, e, nMH);
  }

  // out = rms_norm(h + mixed)
  resid_rms_kernel<<<M_, 256, 0, stream>>>(h, mixed, out, (__bf16*)nullptr, H_);
}
